// MPNN_nk_56229711839360
// MI455X (gfx1250) — compile-verified, matrix operand order checked by reading, not run on hardware
//
#include <hip/hip_runtime.h>

#define NNODES 50000
#define NEDGES 800000
#define DF     128

typedef __attribute__((ext_vector_type(16))) __bf16 v16bf;
typedef __attribute__((ext_vector_type(8)))  __bf16 v8bf;
typedef __attribute__((ext_vector_type(8)))  float  v8f;

// f32 -> bf16 via native conversion hardware (RNE), instead of bit-twiddling.
__device__ __forceinline__ __bf16 f2bf(float f) {
  return (__bf16)f;
}

// ---------------------------------------------------------------------------
// Phase 0: zero the neighbor-sum accumulator (25.6 MB in d_ws)
// ---------------------------------------------------------------------------
__global__ void __launch_bounds__(256) mpnn_zero_ws(float4* __restrict__ p) {
  unsigned i = blockIdx.x * 256u + threadIdx.x;
  p[i] = make_float4(0.f, 0.f, 0.f, 0.f);
}

// ---------------------------------------------------------------------------
// Phase 1: symmetric edge scatter. 32 lanes per edge, 4 floats per lane.
// Accumulator is L2-resident; fire-and-forget global_atomic_add_f32.
// ---------------------------------------------------------------------------
__global__ void __launch_bounds__(256) mpnn_edge_scatter(
    const float* __restrict__ fea, const int* __restrict__ eidx,
    float* __restrict__ nei) {
  unsigned t    = blockIdx.x * 256u + threadIdx.x;
  unsigned e    = t >> 5;
  unsigned lane = t & 31u;
  if (e >= NEDGES) return;
  int s = eidx[e];
  int d = eidx[NEDGES + e];
  unsigned c = lane * 4u;
  const float4 fs = *(const float4*)(fea + (size_t)s * DF + c);
  const float4 fd = *(const float4*)(fea + (size_t)d * DF + c);
  float* pd = nei + (size_t)d * DF + c;
  float* ps = nei + (size_t)s * DF + c;
  atomicAdd(pd + 0, fs.x); atomicAdd(pd + 1, fs.y);
  atomicAdd(pd + 2, fs.z); atomicAdd(pd + 3, fs.w);
  atomicAdd(ps + 0, fd.x); atomicAdd(ps + 1, fd.y);
  atomicAdd(ps + 2, fd.z); atomicAdd(ps + 3, fd.w);
}

// ---------------------------------------------------------------------------
// Phase 2: fused dual GEMM with bias via v_wmma_f32_16x16x32_bf16.
// out = fea @ Wself^T + nei @ Wneig^T + (bself + bneig)
// Block = 128 threads (4 waves); wave owns a 16-row x 128-col output tile.
// Weights staged in LDS as bf16, rows padded to 136 elems (272 B) so the
// 16-byte B-fragment loads from 16 consecutive rows hit distinct bank groups.
// ---------------------------------------------------------------------------
__global__ void __launch_bounds__(128) mpnn_gemm(
    const float* __restrict__ fea, const float* __restrict__ nei,
    const float* __restrict__ Wself, const float* __restrict__ bself,
    const float* __restrict__ Wneig, const float* __restrict__ bneig,
    float* __restrict__ out) {
  __shared__ __bf16 ldsW[2][DF][136];

  // --- stage both weight matrices into LDS (coalesced f32 reads, bf16 stores)
  {
    const float* Wm[2] = {Wself, Wneig};
#pragma unroll
    for (int mat = 0; mat < 2; ++mat) {
      const float* W = Wm[mat];
      for (int i = threadIdx.x * 4; i < DF * DF; i += 128 * 4) {
        float4 f = *(const float4*)(W + i);
        int r = i >> 7, c = i & 127;
        ldsW[mat][r][c + 0] = f2bf(f.x);
        ldsW[mat][r][c + 1] = f2bf(f.y);
        ldsW[mat][r][c + 2] = f2bf(f.z);
        ldsW[mat][r][c + 3] = f2bf(f.w);
      }
    }
  }
  __syncthreads();

  const int wave  = threadIdx.x >> 5;
  const int lane  = threadIdx.x & 31;
  const int half  = lane >> 4;     // which half-wave
  const int l16   = lane & 15;
  const int m0    = (blockIdx.x * 4 + wave) * 16;
  const int kbase = half * 8;      // ISA 16-bit A/B layout: hi half-wave owns K+8

  int row = m0 + l16;              // clamp loads for tail tile (stores masked)
  if (row >= NNODES) row = NNODES - 1;

  // accumulators: 8 column tiles; init with fused bias (constant per lane: N = l16)
  v8f acc[8];
#pragma unroll
  for (int nt = 0; nt < 8; ++nt) {
    float b = bself[nt * 16 + l16] + bneig[nt * 16 + l16];
#pragma unroll
    for (int v = 0; v < 8; ++v) acc[nt][v] = b;
  }

  // K = 256 fused: chunks 0..3 = fea x Wself, chunks 4..7 = nei x Wneig
#pragma unroll
  for (int chunk = 0; chunk < 8; ++chunk) {
    const float* src = (chunk < 4) ? fea : nei;
    const int    mat = (chunk < 4) ? 0 : 1;
    const int    k0  = (chunk & 3) * 32;

    // A fragment: lane's row, K runs [kb..kb+7] and [kb+16..kb+23]
    const float* ap = src + (size_t)row * DF + k0 + kbase;
    float4 a0 = *(const float4*)(ap + 0);
    float4 a1 = *(const float4*)(ap + 4);
    float4 a2 = *(const float4*)(ap + 16);
    float4 a3 = *(const float4*)(ap + 20);
    v16bf a;
    a[0]  = f2bf(a0.x); a[1]  = f2bf(a0.y); a[2]  = f2bf(a0.z); a[3]  = f2bf(a0.w);
    a[4]  = f2bf(a1.x); a[5]  = f2bf(a1.y); a[6]  = f2bf(a1.z); a[7]  = f2bf(a1.w);
    a[8]  = f2bf(a2.x); a[9]  = f2bf(a2.y); a[10] = f2bf(a2.z); a[11] = f2bf(a2.w);
    a[12] = f2bf(a3.x); a[13] = f2bf(a3.y); a[14] = f2bf(a3.z); a[15] = f2bf(a3.w);

#pragma unroll
    for (int nt = 0; nt < 8; ++nt) {
      // B fragment: lane n=l16 holds weight row (nt*16+l16), same K runs as A
      const __bf16* bp = &ldsW[mat][nt * 16 + l16][k0 + kbase];
      v8bf blo = *(const v8bf*)(bp);        // 16B -> ds_load_b128
      v8bf bhi = *(const v8bf*)(bp + 16);
      v16bf bfrag = __builtin_shufflevector(
          blo, bhi, 0, 1, 2, 3, 4, 5, 6, 7, 8, 9, 10, 11, 12, 13, 14, 15);
      acc[nt] = __builtin_amdgcn_wmma_f32_16x16x32_bf16(
          false, a, false, bfrag, (short)0, acc[nt], false, false);
    }
  }

  // D layout: lane holds column N = l16; VGPR v -> row m0 + v + 8*half
#pragma unroll
  for (int nt = 0; nt < 8; ++nt) {
#pragma unroll
    for (int v = 0; v < 8; ++v) {
      int r = m0 + half * 8 + v;
      if (r < NNODES) out[(size_t)r * DF + nt * 16 + l16] = acc[nt][v];
    }
  }
}

// ---------------------------------------------------------------------------
extern "C" void kernel_launch(void* const* d_in, const int* in_sizes, int n_in,
                              void* d_out, int out_size, void* d_ws, size_t ws_size,
                              hipStream_t stream) {
  const float* fea   = (const float*)d_in[0];
  const int*   eidx  = (const int*)d_in[1];
  const float* Wself = (const float*)d_in[2];
  const float* bself = (const float*)d_in[3];
  const float* Wneig = (const float*)d_in[4];
  const float* bneig = (const float*)d_in[5];
  float* out = (float*)d_out;
  float* nei = (float*)d_ws;  // N_NODES * 128 f32 = 25.6 MB scratch

  mpnn_zero_ws<<<(NNODES * DF / 4) / 256, 256, 0, stream>>>((float4*)nei);
  mpnn_edge_scatter<<<(NEDGES * 32) / 256, 256, 0, stream>>>(fea, eidx, nei);
  mpnn_gemm<<<(NNODES + 63) / 64, 128, 0, stream>>>(fea, nei, Wself, bself,
                                                    Wneig, bneig, out);
}
